// NaylisAttention_55344948576747
// MI455X (gfx1250) — matrix operand order repaired and checked by reading, not compiled, not run on hardware
//
#include <hip/hip_runtime.h>
#include <hip/hip_fp16.h>

typedef __attribute__((ext_vector_type(16))) _Float16 v16h;
typedef __attribute__((ext_vector_type(8)))  _Float16 v8h;
typedef __attribute__((ext_vector_type(8)))  float    v8f;

#define S_LEN  2048
#define EMB    2048
#define NH     16
#define NKVH   4
#define RK     32
#define HD     128
#define KV_DIM (NKVH * HD)   // 512
#define HR     (NH * RK)     // 512
#define N_REP  (NH / NKVH)   // 4
#define EPS_RMS 1e-6f

union C8  { v8f v;  float f[8]; };
union A16 { v16h v; v8h h[2]; _Float16 e[16]; };

static __device__ __forceinline__ v8f wmma16(v16h a, v16h b, v8f c) {
  // D = A(16x32 f16) x B(32x16 f16) + C(16x16 f32)
  return __builtin_amdgcn_wmma_f32_16x16x32_f16(false, a, false, b, (short)0, c,
                                                false, false);
}

// ---------------------------------------------------------- async staging ---
// gfx1250 GLOBAL_LOAD_ASYNC_TO_LDS_B128 (ASYNCcnt-tracked).  Probe-confirmed
// signature: (int4 addrspace(1)* gsrc, int4 addrspace(3)* ldst, imm off, imm cpol)
#if defined(__gfx1250__) && __has_builtin(__builtin_amdgcn_global_load_async_to_lds_b128)
#define HAVE_ASYNC_LDS 1
typedef int i32x4 __attribute__((vector_size(16)));
typedef __attribute__((address_space(1))) i32x4 as1_i32x4;
typedef __attribute__((address_space(3))) i32x4 as3_i32x4;
#endif

static __device__ __forceinline__ void stage32B(const _Float16* g, _Float16* l) {
#if defined(HAVE_ASYNC_LDS)
  __builtin_amdgcn_global_load_async_to_lds_b128((as1_i32x4*)(void*)g,
                                                 (as3_i32x4*)(void*)l, 0, 0);
  __builtin_amdgcn_global_load_async_to_lds_b128((as1_i32x4*)(void*)(g + 8),
                                                 (as3_i32x4*)(void*)(l + 8), 0, 0);
#else
  *(v8h*)l       = *(const v8h*)g;
  *(v8h*)(l + 8) = *(const v8h*)(g + 8);
#endif
}
static __device__ __forceinline__ void async_stage_wait() {
#if defined(HAVE_ASYNC_LDS)
  asm volatile("s_wait_asynccnt 0x0" ::: "memory");
#endif
}

// ---------------------------------------------------------------- converts ---
__global__ void cvt_half(const float* __restrict__ in, _Float16* __restrict__ out,
                         long n, float scale) {
  long i = (long)blockIdx.x * blockDim.x + threadIdx.x;
  long stride = (long)gridDim.x * blockDim.x;
  for (; i < n; i += stride) out[i] = (_Float16)(in[i] * scale);
}

// Wrq is [H*R, E]; fold graph_scale[h] into row h*R+r
__global__ void cvt_wrq(const float* __restrict__ in, const float* __restrict__ gs,
                        _Float16* __restrict__ out, long n) {
  long i = (long)blockIdx.x * blockDim.x + threadIdx.x;
  long stride = (long)gridDim.x * blockDim.x;
  for (; i < n; i += stride) {
    int row = (int)(i / EMB);
    out[i] = (_Float16)(in[i] * gs[row / RK]);
  }
}

// ---------------------------------------------------------------- GEMM (NT) ---
// C[M,N] = A[M,K] * B[N,K]^T.  Block = 256 threads = 8 waves, block tile
// 128x128, K-step 32.  A/B panels staged in LDS via async loads; each wave
// computes a 32x64 register tile = 2x4 WMMA accumulators.
#define BM 128
#define BN 128
#define KS 32

__global__ __launch_bounds__(256) void gemm_nt(const _Float16* __restrict__ A,
                                               const _Float16* __restrict__ B,
                                               float* __restrict__ Cf,
                                               _Float16* __restrict__ Ch,
                                               int M, int N, int K) {
  __shared__ _Float16 Ash[BM][KS];   // (m, k) 8 KB
  __shared__ _Float16 Bsh[BN][KS];   // (n, k) 8 KB
  int tid  = threadIdx.x;
  int lane = tid & 31, wave = tid >> 5;
  int wm = wave >> 1, wn = wave & 1;          // 4 x 2 wave grid
  int ml = lane & 15, hb = lane >> 4;         // lane half 0/1
  int m0 = blockIdx.y * BM, n0 = blockIdx.x * BN;

  int srow = tid >> 1;                        // 0..127 staging row
  int scol = (tid & 1) * 16;                  // 0 / 16 halfs
  const _Float16* ag = A + (size_t)(m0 + srow) * K + scol;
  const _Float16* bg = B + (size_t)(n0 + srow) * K + scol;

  v8f zero = {};
  v8f acc[2][4];
  for (int i = 0; i < 2; ++i)
    for (int j = 0; j < 4; ++j) acc[i][j] = zero;

  for (int k0 = 0; k0 < K; k0 += KS) {
    stage32B(ag + k0, &Ash[srow][scol]);
    stage32B(bg + k0, &Bsh[srow][scol]);
    if (k0 + KS < K) {
      __builtin_prefetch(ag + k0 + KS, 0, 0);   // global_prefetch_b8
      __builtin_prefetch(bg + k0 + KS, 0, 0);
    }
    async_stage_wait();
    __syncthreads();

    A16 af[2], bf[4];
    for (int mt = 0; mt < 2; ++mt) {
      int row = wm * 32 + mt * 16 + ml;
      af[mt].h[0] = *(const v8h*)&Ash[row][hb * 8];        // K = hb*8 + 0..7
      af[mt].h[1] = *(const v8h*)&Ash[row][16 + hb * 8];   // K = 16 + hb*8 + 0..7
    }
    for (int nt = 0; nt < 4; ++nt) {
      int n = wn * 64 + nt * 16 + ml;
      bf[nt].h[0] = *(const v8h*)&Bsh[n][hb * 8];          // K = hb*8 + 0..7
      bf[nt].h[1] = *(const v8h*)&Bsh[n][16 + hb * 8];     // K = 16 + hb*8 + 0..7
    }
    for (int mt = 0; mt < 2; ++mt)
      for (int nt = 0; nt < 4; ++nt)
        acc[mt][nt] = wmma16(af[mt].v, bf[nt].v, acc[mt][nt]);
    __syncthreads();
  }

  for (int mt = 0; mt < 2; ++mt)
    for (int nt = 0; nt < 4; ++nt) {
      C8 cc; cc.v = acc[mt][nt];
      int rbase = m0 + wm * 32 + mt * 16 + 8 * hb;
      int cbase = n0 + wn * 64 + nt * 16 + ml;
      if (Cf) {
        for (int r = 0; r < 8; ++r)
          Cf[(size_t)(rbase + r) * N + cbase] = cc.f[r];
      }
      if (Ch) {
        for (int r = 0; r < 8; ++r)
          Ch[(size_t)(rbase + r) * N + cbase] = (_Float16)cc.f[r];
      }
    }
}

// --------------------------------------------------- RMSNorm + RoPE + cvt ---
__global__ __launch_bounds__(32) void rmsnorm_rope(const float* __restrict__ in,
                                                   const float* __restrict__ w,
                                                   _Float16* __restrict__ out,
                                                   int nheads, float outscale) {
  int s = blockIdx.y, hh = blockIdx.x, lane = threadIdx.x;
  int rowdim = nheads * HD;
  const float* base = in + (size_t)s * rowdim + hh * HD;
  float x[4], ss = 0.f;
  for (int i = 0; i < 4; ++i) { x[i] = base[lane + 32 * i]; ss += x[i] * x[i]; }
  for (int m = 1; m < 32; m <<= 1) ss += __shfl_xor(ss, m, 32);
  float r = rsqrtf(ss * (1.0f / HD) + EPS_RMS);
  float nx[4];
  for (int i = 0; i < 4; ++i) nx[i] = x[i] * r * w[lane + 32 * i];
  _Float16* ob = out + (size_t)s * rowdim + hh * HD;
  for (int i = 0; i < 4; ++i) {
    int d = lane + 32 * i;
    int j = d & 63;
    float ang = (float)s * powf(10000.f, -(float)j * (1.0f / 64.f));
    float rot = (i < 2) ? -nx[i + 2] : nx[i - 2];
    ob[d] = (_Float16)((nx[i] * cosf(ang) + rot * sinf(ang)) * outscale);
  }
}

// ------------------------------------------------------- flash attention ----
__global__ __launch_bounds__(32) void attn_fwd(
    const _Float16* __restrict__ qh,   // [S, E]  (pre-scaled by 1/sqrt(HD))
    const _Float16* __restrict__ kh,   // [S, KV_DIM]
    const _Float16* __restrict__ vh,   // [S, KV_DIM]
    const _Float16* __restrict__ rqh,  // [S, H*R] (graph_scale folded)
    const _Float16* __restrict__ rkh,  // [S, H*R]
    float* __restrict__ o) {           // [S, E]
  __shared__ _Float16 v_lds[32][HD];   // V block, row-major (key, d)
  __shared__ _Float16 p_lds[16][32];   // probabilities tile, row-major (m, key)
  int lane = threadIdx.x;
  int qi = blockIdx.x, h = blockIdx.y;
  int kvh = h / N_REP;
  int m0 = qi << 4;
  int ml = lane & 15, hb = lane >> 4;

  const _Float16* qrow = qh + (size_t)(m0 + ml) * EMB + h * HD;
  A16 qf[4];
  for (int c = 0; c < 4; ++c) {
    qf[c].h[0] = *(const v8h*)(qrow + 32 * c + hb * 8);
    qf[c].h[1] = *(const v8h*)(qrow + 32 * c + 16 + hb * 8);
  }
  const _Float16* rqrow = rqh + (size_t)(m0 + ml) * HR + h * RK;
  A16 rqf;
  rqf.h[0] = *(const v8h*)(rqrow + hb * 8);
  rqf.h[1] = *(const v8h*)(rqrow + 16 + hb * 8);

  C8 acc[8];
  float mrow[8], lrow[8];
  for (int c = 0; c < 8; ++c)
    for (int r = 0; r < 8; ++r) acc[c].f[r] = 0.f;
  for (int r = 0; r < 8; ++r) { mrow[r] = -1e30f; lrow[r] = 0.f; }

  int kbmax = (m0 + 15) >> 5;
  for (int kb = 0; kb <= kbmax; ++kb) {
    int n0 = kb << 5;
    { // stage V block rows into LDS (lane -> key row, 256 B each)
      const _Float16* src = vh + (size_t)(n0 + lane) * KV_DIM + kvh * HD;
      _Float16* dst = &v_lds[lane][0];
      for (int i = 0; i < 8; ++i) stage32B(src + 16 * i, dst + 16 * i);
      async_stage_wait();
    }
    if (kb < kbmax)
      __builtin_prefetch(kh + (size_t)(n0 + 32 + lane) * KV_DIM + kvh * HD, 0, 0);

    // scores: s0 = keys n0..n0+15, s1 = keys n0+16..n0+31
    v8f sv0 = {}, sv1 = {};
    const _Float16* kr0 = kh + (size_t)(n0 + ml) * KV_DIM + kvh * HD;
    const _Float16* kr1 = kh + (size_t)(n0 + 16 + ml) * KV_DIM + kvh * HD;
    for (int c = 0; c < 4; ++c) {
      A16 b0, b1;
      b0.h[0] = *(const v8h*)(kr0 + 32 * c + hb * 8);
      b0.h[1] = *(const v8h*)(kr0 + 32 * c + 16 + hb * 8);
      b1.h[0] = *(const v8h*)(kr1 + 32 * c + hb * 8);
      b1.h[1] = *(const v8h*)(kr1 + 32 * c + 16 + hb * 8);
      sv0 = wmma16(qf[c].v, b0.v, sv0);
      sv1 = wmma16(qf[c].v, b1.v, sv1);
    }
    { // low-rank graph bias
      const _Float16* rr0 = rkh + (size_t)(n0 + ml) * HR + h * RK;
      const _Float16* rr1 = rkh + (size_t)(n0 + 16 + ml) * HR + h * RK;
      A16 b0, b1;
      b0.h[0] = *(const v8h*)(rr0 + hb * 8); b0.h[1] = *(const v8h*)(rr0 + 16 + hb * 8);
      b1.h[0] = *(const v8h*)(rr1 + hb * 8); b1.h[1] = *(const v8h*)(rr1 + 16 + hb * 8);
      sv0 = wmma16(rqf.v, b0.v, sv0);
      sv1 = wmma16(rqf.v, b1.v, sv1);
    }
    C8 s0, s1; s0.v = sv0; s1.v = sv1;

    // causal mask + online softmax (rows r+8*hb live in this lane half)
    for (int r = 0; r < 8; ++r) {
      int mg = m0 + r + 8 * hb;
      if (n0 + ml > mg)      s0.f[r] = -1e30f;
      if (n0 + 16 + ml > mg) s1.f[r] = -1e30f;
      float cur = fmaxf(s0.f[r], s1.f[r]);
      for (int m = 1; m < 16; m <<= 1) cur = fmaxf(cur, __shfl_xor(cur, m, 32));
      float mn = fmaxf(mrow[r], cur);
      float alpha = __expf(mrow[r] - mn);
      float p0 = __expf(s0.f[r] - mn);
      float p1 = __expf(s1.f[r] - mn);
      float rs = p0 + p1;
      for (int m = 1; m < 16; m <<= 1) rs += __shfl_xor(rs, m, 32);
      lrow[r] = lrow[r] * alpha + rs;
      mrow[r] = mn;
      for (int c = 0; c < 8; ++c) acc[c].f[r] *= alpha;
      p_lds[r + 8 * hb][ml]      = (_Float16)p0;
      p_lds[r + 8 * hb][16 + ml] = (_Float16)p1;
    }

    // reload P as an A fragment (16x32, K = key-in-block)
    A16 pf;
    {
      const v8h* pr = (const v8h*)&p_lds[ml][0];
      pf.h[0] = pr[hb];
      pf.h[1] = pr[2 + hb];
    }
    // acc += P(16x32) * V(32x128)
    for (int c2 = 0; c2 < 8; ++c2) {
      A16 vf;
      for (int j = 0; j < 16; ++j) vf.e[j] = v_lds[(j < 8 ? hb * 8 + j : 16 + hb * 8 + j - 8)][c2 * 16 + ml];
      acc[c2].v = wmma16(pf.v, vf.v, acc[c2].v);
    }
  }

  for (int c2 = 0; c2 < 8; ++c2)
    for (int r = 0; r < 8; ++r)
      o[(size_t)(m0 + r + 8 * hb) * EMB + h * HD + c2 * 16 + ml] =
          acc[c2].f[r] / lrow[r];
}

// ------------------------------------------------------------------ driver ---
extern "C" void kernel_launch(void* const* d_in, const int* in_sizes, int n_in,
                              void* d_out, int out_size, void* d_ws, size_t ws_size,
                              hipStream_t stream) {
  const float* x   = (const float*)d_in[0];
  const float* Wq  = (const float*)d_in[1];
  const float* Wk  = (const float*)d_in[2];
  const float* Wv  = (const float*)d_in[3];
  const float* Wo  = (const float*)d_in[4];
  const float* Wrq = (const float*)d_in[5];
  const float* Wrk = (const float*)d_in[6];
  const float* gs  = (const float*)d_in[7];
  const float* qw  = (const float*)d_in[8];
  const float* kw  = (const float*)d_in[9];

  char* ws = (char*)d_ws;
  size_t off = 0;
  auto alloc = [&](size_t bytes) -> char* {
    char* p = ws + off;
    off = (off + bytes + 255) & ~(size_t)255;
    return p;
  };
  _Float16* xh   = (_Float16*)alloc((size_t)S_LEN * EMB * 2);
  _Float16* wqh  = (_Float16*)alloc((size_t)EMB * EMB * 2);
  _Float16* wkh  = (_Float16*)alloc((size_t)KV_DIM * EMB * 2);
  _Float16* wvh  = (_Float16*)alloc((size_t)KV_DIM * EMB * 2);
  _Float16* woh  = (_Float16*)alloc((size_t)EMB * EMB * 2);
  _Float16* wrqh = (_Float16*)alloc((size_t)HR * EMB * 2);
  _Float16* wrkh = (_Float16*)alloc((size_t)HR * EMB * 2);
  float*    qf   = (float*)   alloc((size_t)S_LEN * EMB * 4);
  float*    kf   = (float*)   alloc((size_t)S_LEN * KV_DIM * 4);
  _Float16* qh2  = (_Float16*)alloc((size_t)S_LEN * EMB * 2);
  _Float16* kh2  = (_Float16*)alloc((size_t)S_LEN * KV_DIM * 2);
  _Float16* vh2  = (_Float16*)alloc((size_t)S_LEN * KV_DIM * 2);
  _Float16* rqh2 = (_Float16*)alloc((size_t)S_LEN * HR * 2);
  _Float16* rkh2 = (_Float16*)alloc((size_t)S_LEN * HR * 2);
  float*    af   = (float*)   alloc((size_t)S_LEN * EMB * 4);
  _Float16* ah   = (_Float16*)alloc((size_t)S_LEN * EMB * 2);

  auto cgrid = [](long n) { long b = (n + 255) / 256; return (int)(b > 4096 ? 4096 : b); };

  cvt_half<<<cgrid((long)S_LEN * EMB), 256, 0, stream>>>(x, xh, (long)S_LEN * EMB, 1.f);
  cvt_half<<<cgrid((long)EMB * EMB), 256, 0, stream>>>(Wq, wqh, (long)EMB * EMB, 1.f);
  cvt_half<<<cgrid((long)KV_DIM * EMB), 256, 0, stream>>>(Wk, wkh, (long)KV_DIM * EMB, 1.f);
  cvt_half<<<cgrid((long)KV_DIM * EMB), 256, 0, stream>>>(Wv, wvh, (long)KV_DIM * EMB, 1.f);
  cvt_half<<<cgrid((long)EMB * EMB), 256, 0, stream>>>(Wo, woh, (long)EMB * EMB, 1.f);
  cvt_half<<<cgrid((long)HR * EMB), 256, 0, stream>>>(Wrk, wrkh, (long)HR * EMB, 1.f);
  cvt_wrq<<<cgrid((long)HR * EMB), 256, 0, stream>>>(Wrq, gs, wrqh, (long)HR * EMB);

  // projections (LDS-tiled WMMA GEMMs)
  gemm_nt<<<dim3(EMB / BN, S_LEN / BM), 256, 0, stream>>>(xh, wqh, qf, nullptr, S_LEN, EMB, EMB);
  gemm_nt<<<dim3(KV_DIM / BN, S_LEN / BM), 256, 0, stream>>>(xh, wkh, kf, nullptr, S_LEN, KV_DIM, EMB);
  gemm_nt<<<dim3(KV_DIM / BN, S_LEN / BM), 256, 0, stream>>>(xh, wvh, nullptr, vh2, S_LEN, KV_DIM, EMB);
  gemm_nt<<<dim3(HR / BN, S_LEN / BM), 256, 0, stream>>>(xh, wrqh, nullptr, rqh2, S_LEN, HR, EMB);
  gemm_nt<<<dim3(HR / BN, S_LEN / BM), 256, 0, stream>>>(xh, wrkh, nullptr, rkh2, S_LEN, HR, EMB);

  // QK rmsnorm + rope; fold 1/sqrt(HD) into Q
  rmsnorm_rope<<<dim3(NH, S_LEN), 32, 0, stream>>>(qf, qw, qh2, NH, 0.08838834764831845f);
  rmsnorm_rope<<<dim3(NKVH, S_LEN), 32, 0, stream>>>(kf, kw, kh2, NKVH, 1.0f);

  // flash attention with graph bias
  attn_fwd<<<dim3(S_LEN / 16, NH), 32, 0, stream>>>(qh2, kh2, vh2, rqh2, rkh2, af);

  // output projection
  cvt_half<<<cgrid((long)S_LEN * EMB), 256, 0, stream>>>(af, ah, (long)S_LEN * EMB, 1.f);
  gemm_nt<<<dim3(EMB / BN, S_LEN / BM), 256, 0, stream>>>(ah, woh, (float*)d_out, nullptr, S_LEN, EMB, EMB);
}
